// SkeletonGraphConv_26663156974180
// MI455X (gfx1250) — matrix operand order repaired and checked by reading, not compiled, not run on hardware
//
#include <hip/hip_runtime.h>
#include <hip/hip_bf16.h>

typedef __attribute__((ext_vector_type(16))) __bf16        v16bf;
typedef __attribute__((ext_vector_type(8)))  float         v8f;
typedef __attribute__((ext_vector_type(8)))  unsigned int  v8u;

#define C_IN   64
#define C_OUT  128
#define BN_EPS 1e-5f
#define LEAK   0.2f
#define WSW_ELEMS 8192            // 64*128 bf16, fragment-swizzled

__device__ __forceinline__ void atomAddF(float* p, float v) {
    unsafeAtomicAdd(p, v);   // global_atomic_add_f32 on gfx1250
}

// ---------------- 1. init deg (self loop) ----------------
__global__ void k_init_deg(float* __restrict__ deg, int N) {
    int i = blockIdx.x * blockDim.x + threadIdx.x;
    if (i < N) deg[i] = 1.0f;
}

// ---------------- 1b. zero BN accumulators ----------------
__global__ void k_zero_stats(float* __restrict__ sums) {
    sums[threadIdx.x] = 0.0f;     // 256 threads: sums[0..127] ++ sumsq[0..127]
}

// ---------------- 2. degree from edges ----------------
__global__ void k_deg_edges(const long long* __restrict__ ei, float* __restrict__ deg,
                            int E) {
    int e = blockIdx.x * blockDim.x + threadIdx.x;
    if (e < E) atomAddF(&deg[ei[(long long)E + e]], 1.0f);
}

// ---------------- 3. dis = deg^-1/2 ----------------
__global__ void k_dis(const float* __restrict__ deg, float* __restrict__ dis, int N) {
    int i = blockIdx.x * blockDim.x + threadIdx.x;
    if (i < N) {
        float d = deg[i];
        dis[i] = (d > 0.0f) ? __frsqrt_rn(fmaxf(d, 1.0f)) : 0.0f;
    }
}

// -------- 3b. swizzle W (f32, KxN row-major) -> bf16 fragment order ----------
// layout: wbf[(((ct*2 + ks)*32 + lane)*16 + e], e=2j+p holds
//         W[k = ks*32 + (e>>3)*16 + (lane>>4)*8 + (e&7)][n = ct*16 + (lane&15)]
__global__ void k_convW(const float* __restrict__ W, __bf16* __restrict__ wbf) {
    int t = blockIdx.x * blockDim.x + threadIdx.x;   // 8192 threads
    if (t < WSW_ELEMS) {
        int e    = t & 15;
        int lane = (t >> 4) & 31;
        int ks   = (t >> 9) & 1;
        int ct   = t >> 10;
        int k = ks * 32 + (e >> 3) * 16 + (lane >> 4) * 8 + (e & 7);
        int n = ct * 16 + (lane & 15);
        wbf[t] = (__bf16)W[k * C_OUT + n];
    }
}

// ---------------- 4. WMMA GEMM: xw = x @ W (bf16 in, f32 acc) ----------------
// block = 256 thr = 8 waves; each wave: 16 rows x all 128 cols (16 WMMAs)
__global__ __launch_bounds__(256) void k_gemm(const float* __restrict__ x,
                                              const __bf16* __restrict__ wbf,
                                              float* __restrict__ xw, int N) {
    __shared__ __bf16 sW[WSW_ELEMS];                 // 16 KB, fragment order

    const int tid  = threadIdx.x;
    const int wave = tid >> 5;
    const int lane = tid & 31;
    const int half = lane >> 4;
    const int mn   = lane & 15;

    // async DMA the pre-swizzled W image into LDS: 16KB = 256 thr * 4 * b128
    {
        unsigned lbase = (unsigned)(size_t)(&sW[0]) + (unsigned)tid * 16u;
        unsigned long long gbase = (unsigned long long)(const void*)wbf
                                   + (unsigned long long)tid * 16ull;
        #pragma unroll
        for (int i = 0; i < 4; ++i) {
            unsigned           l = lbase + (unsigned)(i * 4096);
            unsigned long long g = gbase + (unsigned long long)(i * 4096);
            asm volatile("global_load_async_to_lds_b128 %0, %1, off"
                         :: "v"(l), "v"(g) : "memory");
        }
        asm volatile("s_wait_asynccnt 0x0" ::: "memory");
    }
    __syncthreads();

    const long long rowBase = (long long)blockIdx.x * 128 + (long long)wave * 16;
    if (rowBase + 16 > (long long)N) return;         // after the only barrier

    // ---- load this lane's A fragments straight from global into registers ----
    // lane covers row m=lane&15; K runs: half*8 + {0,16,32,48}, 8 floats each
    const float* rowp = x + (rowBase + mn) * C_IN + half * 8;
    v16bf a0, a1;
    {
        float4 f0 = *(const float4*)(rowp + 0);
        float4 f1 = *(const float4*)(rowp + 4);
        float4 f2 = *(const float4*)(rowp + 16);
        float4 f3 = *(const float4*)(rowp + 20);
        float4 f4 = *(const float4*)(rowp + 32);
        float4 f5 = *(const float4*)(rowp + 36);
        float4 f6 = *(const float4*)(rowp + 48);
        float4 f7 = *(const float4*)(rowp + 52);
        a0[0]=(__bf16)f0.x;  a0[1]=(__bf16)f0.y;  a0[2]=(__bf16)f0.z;  a0[3]=(__bf16)f0.w;
        a0[4]=(__bf16)f1.x;  a0[5]=(__bf16)f1.y;  a0[6]=(__bf16)f1.z;  a0[7]=(__bf16)f1.w;
        a0[8]=(__bf16)f2.x;  a0[9]=(__bf16)f2.y;  a0[10]=(__bf16)f2.z; a0[11]=(__bf16)f2.w;
        a0[12]=(__bf16)f3.x; a0[13]=(__bf16)f3.y; a0[14]=(__bf16)f3.z; a0[15]=(__bf16)f3.w;
        a1[0]=(__bf16)f4.x;  a1[1]=(__bf16)f4.y;  a1[2]=(__bf16)f4.z;  a1[3]=(__bf16)f4.w;
        a1[4]=(__bf16)f5.x;  a1[5]=(__bf16)f5.y;  a1[6]=(__bf16)f5.z;  a1[7]=(__bf16)f5.w;
        a1[8]=(__bf16)f6.x;  a1[9]=(__bf16)f6.y;  a1[10]=(__bf16)f6.z; a1[11]=(__bf16)f6.w;
        a1[12]=(__bf16)f7.x; a1[13]=(__bf16)f7.y; a1[14]=(__bf16)f7.z; a1[15]=(__bf16)f7.w;
    }

    // ---- sweep all 8 column tiles, reusing A fragments ----
    #pragma unroll
    for (int ct = 0; ct < 8; ++ct) {
        const v8u* bp0 = (const v8u*)&sW[(((ct << 1) | 0) * 32 + lane) * 16];
        const v8u* bp1 = (const v8u*)&sW[(((ct << 1) | 1) * 32 + lane) * 16];
        v16bf b0 = __builtin_bit_cast(v16bf, *bp0);   // 2x ds_load_b128
        v16bf b1 = __builtin_bit_cast(v16bf, *bp1);
        v8f acc = {};
        acc = __builtin_amdgcn_wmma_f32_16x16x32_bf16(false, a0, false, b0,
                                                      (short)0, acc, false, false);
        acc = __builtin_amdgcn_wmma_f32_16x16x32_bf16(false, a1, false, b1,
                                                      (short)0, acc, false, false);
        // C layout: lane<16 -> rows 0..7, lane>=16 -> rows 8..15
        float* outp = xw + (rowBase + half * 8) * C_OUT + ct * 16 + mn;
        #pragma unroll
        for (int r = 0; r < 8; ++r) outp[(long long)r * C_OUT] = acc[r];
    }
}

// ---------------- 5. out = xw * dis^2 + b  (self-loop + bias) ----------------
__global__ void k_self(const float* __restrict__ xw, const float* __restrict__ dis,
                       const float* __restrict__ bias, float* __restrict__ out,
                       long long total) {
    long long g = (long long)blockIdx.x * blockDim.x + threadIdx.x;
    if (g < total) {
        long long i = g >> 7;
        int       c = (int)(g & 127);
        float s = dis[i];
        out[g] = xw[g] * s * s + bias[c];
    }
}

// ---------------- 6. edge scatter: out[dst] += xw[src]*norm ----------------
__global__ __launch_bounds__(128) void k_scatter(const long long* __restrict__ ei,
                                                 const float* __restrict__ xw,
                                                 const float* __restrict__ dis,
                                                 float* __restrict__ out, int E) {
    int e = blockIdx.x;
    int c = threadIdx.x;
    long long s = ei[e];
    long long d = ei[(long long)E + e];
    float nrm = dis[s] * dis[d];
    atomAddF(&out[d * C_OUT + c], xw[s * C_OUT + c] * nrm);
}

// ---------------- 7. BN stats ----------------
__global__ __launch_bounds__(128) void k_stats(const float* __restrict__ out,
                                               float* __restrict__ sums,
                                               float* __restrict__ sumsq, int N) {
    int c = threadIdx.x;
    float s = 0.0f, sq = 0.0f;
    for (long long i = blockIdx.x; i < N; i += gridDim.x) {
        float v = out[i * C_OUT + c];
        s += v;
        sq += v * v;
    }
    atomAddF(&sums[c], s);
    atomAddF(&sumsq[c], sq);
}

// ---------------- 8. finalize: BN + LeakyReLU (in place) ----------------
__global__ void k_final(float* __restrict__ out, const float* __restrict__ sums,
                        const float* __restrict__ sumsq,
                        const float* __restrict__ gamma, const float* __restrict__ beta,
                        int N, long long total) {
    long long g = (long long)blockIdx.x * blockDim.x + threadIdx.x;
    if (g < total) {
        int c = (int)(g & 127);
        float invN = 1.0f / (float)N;
        float mean = sums[c] * invN;
        float var  = fmaxf(sumsq[c] * invN - mean * mean, 0.0f);
        float inv  = __frsqrt_rn(var + BN_EPS);
        float y    = gamma[c] * (out[g] - mean) * inv + beta[c];
        out[g] = (y > 0.0f) ? y : LEAK * y;
    }
}

extern "C" void kernel_launch(void* const* d_in, const int* in_sizes, int n_in,
                              void* d_out, int out_size, void* d_ws, size_t ws_size,
                              hipStream_t stream) {
    const float*     x     = (const float*)d_in[0];
    const long long* ei    = (const long long*)d_in[1];
    const float*     W     = (const float*)d_in[5];
    const float*     bias  = (const float*)d_in[6];
    const float*     gamma = (const float*)d_in[7];
    const float*     beta  = (const float*)d_in[8];
    float*           out   = (float*)d_out;

    const int N = in_sizes[0] / C_IN;     // 540672
    const int E = in_sizes[1] / 2;        // 600000
    const long long total = (long long)N * C_OUT;

    // workspace layout (floats then bf16 tail)
    float*  xw    = (float*)d_ws;         // N*128
    float*  deg   = xw + total;           // N
    float*  dis   = deg + N;              // N
    float*  sums  = dis + N;              // 128
    float*  sumsq = sums + C_OUT;         // 128
    __bf16* wbf   = (__bf16*)(sumsq + C_OUT);  // 8192 bf16 (16 KB), 16B-aligned

    const int T = 256;
    k_init_deg  <<<(N + T - 1) / T, T, 0, stream>>>(deg, N);
    k_zero_stats<<<1, 256, 0, stream>>>(sums);
    k_deg_edges <<<(E + T - 1) / T, T, 0, stream>>>(ei, deg, E);
    k_dis       <<<(N + T - 1) / T, T, 0, stream>>>(deg, dis, N);
    k_convW     <<<WSW_ELEMS / T, T, 0, stream>>>(W, wbf);

    int gemmBlocks = (N + 127) / 128;     // 128 rows per block
    k_gemm<<<gemmBlocks, 256, 0, stream>>>(x, wbf, xw, N);

    k_self   <<<(int)((total + T - 1) / T), T, 0, stream>>>(xw, dis, bias, out, total);
    k_scatter<<<E, 128, 0, stream>>>(ei, xw, dis, out, E);
    k_stats  <<<1024, 128, 0, stream>>>(out, sums, sumsq, N);
    k_final  <<<(int)((total + T - 1) / T), T, 0, stream>>>(out, sums, sumsq,
                                                            gamma, beta, N, total);
}
